// CrossLoss_2388001817239
// MI455X (gfx1250) — matrix operand order repaired and checked by reading, not compiled, not run on hardware
//
#include <hip/hip_runtime.h>

// NT-Xent loss, fused:  loss_i = log(sum_{j!=i} exp(sim_ij/T)) - sim_{i,partner}/T
// sim = n n^T via v_wmma_f32_16x16x32_f16; A/B tiles staged in LDS by the
// CDNA5 Tensor Data Mover (tensor_load_to_lds, TENSORcnt double buffering).

#define NROWS 8192            // 2B
#define DDIM  512
#define INV_T 2.0f            // 1 / 0.5
#define BM    128             // block tile rows
#define BN    128             // block tile cols
#define BK    32              // K step

typedef __attribute__((ext_vector_type(16))) _Float16 v16h;
typedef __attribute__((ext_vector_type(8)))  _Float16 v8h;
typedef __attribute__((ext_vector_type(8)))  float    v8f;
typedef __attribute__((ext_vector_type(4)))  unsigned int v4u;
typedef __attribute__((ext_vector_type(8)))  int      v8i;
typedef __attribute__((ext_vector_type(4)))  int      v4i;

// ---------------- 1) row-normalize to f16 ----------------
__global__ void normalize_kernel(const float* __restrict__ x1, const float* __restrict__ x2,
                                 const float* __restrict__ z1, const float* __restrict__ z2,
                                 _Float16* __restrict__ Nh) {
  int row = blockIdx.x;                 // 0..16383 (half*8192 + i)
  int h   = row >> 13;
  int i   = row & (NROWS - 1);
  const float* src;
  if (h == 0) src = (i < 4096) ? (x1 + (size_t)i * DDIM) : (z2 + (size_t)(i - 4096) * DDIM);
  else        src = (i < 4096) ? (x2 + (size_t)i * DDIM) : (z1 + (size_t)(i - 4096) * DDIM);
  int t = threadIdx.x;                  // 256 threads, 2 elems each
  float a0 = src[2 * t], a1 = src[2 * t + 1];
  __shared__ float red[256];
  red[t] = a0 * a0 + a1 * a1;
  __syncthreads();
  for (int s = 128; s > 0; s >>= 1) { if (t < s) red[t] += red[t + s]; __syncthreads(); }
  float inv = rsqrtf(red[0]);
  _Float16* dst = Nh + (size_t)row * DDIM;
  dst[2 * t]     = (_Float16)(a0 * inv);
  dst[2 * t + 1] = (_Float16)(a1 * inv);
}

// ---------------- 2) positive-pair dots ----------------
__global__ void pos_kernel(const _Float16* __restrict__ Nh, float* __restrict__ pos) {
  int row = blockIdx.x;
  int h = row >> 13;
  int i = row & (NROWS - 1);
  int partner = (i + 4096) & (NROWS - 1);
  const _Float16* a = Nh + (size_t)(h * NROWS + i) * DDIM;
  const _Float16* b = Nh + (size_t)(h * NROWS + partner) * DDIM;
  int t = threadIdx.x;
  float s = (float)a[2 * t] * (float)b[2 * t] + (float)a[2 * t + 1] * (float)b[2 * t + 1];
  __shared__ float red[256];
  red[t] = s;
  __syncthreads();
  for (int k = 128; k > 0; k >>= 1) { if (t < k) red[t] += red[t + k]; __syncthreads(); }
  if (t == 0) pos[row] = red[0];
}

// ---------------- TDM: DMA a [rows x 32]-f16 tile (row stride 512) into LDS ----
#if __has_builtin(__builtin_amdgcn_tensor_load_to_lds) && \
    __has_builtin(__builtin_amdgcn_s_wait_tensorcnt)
#define USE_TDM 1
__device__ __forceinline__ void tdm_load_tile(unsigned lds_off, const _Float16* gptr, int rows) {
  unsigned long long ga = (unsigned long long)(uintptr_t)gptr;
  v4u g0;
  g0.x = 1u;                                              // count=1, user descriptor
  g0.y = lds_off;                                         // LDS byte address
  g0.z = (unsigned)(ga & 0xFFFFFFFFu);                    // global_addr[31:0]
  g0.w = (unsigned)((ga >> 32) & 0x1FFFFFFu) | (2u << 30); // global_addr[56:32] | type=2
  v8i g1;
  g1[0] = (int)(1u << 16);                                // wg_mask=0, data_size=1 (2B)
  g1[1] = (int)((DDIM & 0xFFFFu) << 16);                  // tensor_dim0[15:0]
  g1[2] = (int)((DDIM >> 16) | ((unsigned)rows << 16));   // tensor_dim0[31:16] | tensor_dim1[15:0]
  g1[3] = (int)(((unsigned)rows >> 16) | ((unsigned)BK << 16)); // tensor_dim1[31:16] | tile_dim0
  g1[4] = (int)(unsigned)rows;                            // tile_dim1 = rows, tile_dim2 = 0
  g1[5] = (int)(unsigned)DDIM;                            // tensor_dim0_stride[31:0]
  g1[6] = 0;                                              // stride0[47:32] | stride1[15:0]
  g1[7] = 0;                                              // stride1[47:16]
  // This toolchain exposes the 6-arg form: (g0, g1, g2, g3, g4, cpol)
  __builtin_amdgcn_tensor_load_to_lds(g0, g1, (v4i)0, (v4i)0, (v8i)0, 0);
}
#else
#define USE_TDM 0
#endif

// ---------------- 3) fused GEMM + exp row-reduction ----------------
// grid: x = col tile (BN), y = row tile (BM), z = half. 256 threads = 8 waves,
// wave (wm,wn) = (wave>>1, wave&1) owns a 32(M) x 64(N) sub-tile of BM x BN.
__global__ void __launch_bounds__(256)
gemm_exp_kernel(const _Float16* __restrict__ NhAll, float* __restrict__ rowsum) {
  __shared__ _Float16 As[2][BM][BK];   // 16 KB
  __shared__ _Float16 Bs[2][BN][BK];   // 16 KB

  int h    = blockIdx.z;
  const _Float16* Nh = NhAll + (size_t)h * NROWS * DDIM;
  int wave = threadIdx.x >> 5;
  int lane = threadIdx.x & 31;
  int wm = wave >> 1, wn = wave & 1;
  int rowTile = blockIdx.y * BM;
  int colTile = blockIdx.x * BN;
  int rowbase = rowTile + wm * 32;     // this wave's global M base
  int colbase = colTile + wn * 64;     // this wave's global N base
  int lp = lane & 15, hi = lane >> 4;

  v8f acc[2][4] = {};                  // 32(M) x 64(N) in 16x16 tiles

#if USE_TDM
  if (wave == 0) {                     // prime buffer 0 via the Tensor Data Mover
    tdm_load_tile((unsigned)(size_t)&As[0][0][0], Nh + (size_t)rowTile * DDIM, BM);
    tdm_load_tile((unsigned)(size_t)&Bs[0][0][0], Nh + (size_t)colTile * DDIM, BN);
  }
#endif

  int cur = 0;
#pragma unroll 1
  for (int kk = 0; kk < DDIM; kk += BK) {
#if USE_TDM
    if (wave == 0) {
      if (kk + BK < DDIM) {            // stream next K-slab into the other buffer
        tdm_load_tile((unsigned)(size_t)&As[cur ^ 1][0][0],
                      Nh + (size_t)rowTile * DDIM + kk + BK, BM);
        tdm_load_tile((unsigned)(size_t)&Bs[cur ^ 1][0][0],
                      Nh + (size_t)colTile * DDIM + kk + BK, BN);
        __builtin_amdgcn_s_wait_tensorcnt(2);  // current buffer's 2 DMAs complete
      } else {
        __builtin_amdgcn_s_wait_tensorcnt(0);
      }
    }
    __syncthreads();                   // LDS tile `cur` visible to all waves
#else
    {                                  // fallback: cooperative 256-thread copy
      const v8h* gA = (const v8h*)(Nh + (size_t)(rowTile + (threadIdx.x >> 1)) * DDIM + kk);
      const v8h* gB = (const v8h*)(Nh + (size_t)(colTile + (threadIdx.x >> 1)) * DDIM + kk);
      int kq = (threadIdx.x & 1) * 2;  // which pair of 8-half chunks
      ((v8h*)&As[cur][threadIdx.x >> 1][0])[kq]     = gA[kq];
      ((v8h*)&As[cur][threadIdx.x >> 1][0])[kq + 1] = gA[kq + 1];
      ((v8h*)&Bs[cur][threadIdx.x >> 1][0])[kq]     = gB[kq];
      ((v8h*)&Bs[cur][threadIdx.x >> 1][0])[kq + 1] = gB[kq + 1];
      __syncthreads();
    }
#endif
    // A 16x32 layout: lane<16 -> M=lane, K chunks [0..7]&[16..23]; lane>=16 -> M=lane-16, [8..15]&[24..31]
    v16h a[2], b[4];
#pragma unroll
    for (int mi = 0; mi < 2; ++mi) {
      union { v16h v; v8h q[2]; } u;
      int lr = wm * 32 + mi * 16 + lp;
      u.q[0] = *(const v8h*)&As[cur][lr][hi * 8];
      u.q[1] = *(const v8h*)&As[cur][lr][16 + hi * 8];
      a[mi] = u.v;
    }
    // B 32x16 layout: lane<16 -> N=lane, K chunks [0..7]&[16..23]; lane>=16 -> N=lane-16, [8..15]&[24..31]
#pragma unroll
    for (int ni = 0; ni < 4; ++ni) {
      union { v16h v; v8h q[2]; } u;
      int lc = wn * 64 + ni * 16 + lp;
      u.q[0] = *(const v8h*)&Bs[cur][lc][hi * 8];
      u.q[1] = *(const v8h*)&Bs[cur][lc][16 + hi * 8];
      b[ni] = u.v;
    }
#pragma unroll
    for (int mi = 0; mi < 2; ++mi)
#pragma unroll
      for (int ni = 0; ni < 4; ++ni)
        acc[mi][ni] = __builtin_amdgcn_wmma_f32_16x16x32_f16(
            false, a[mi], false, b[ni], (short)0, acc[mi][ni], false, false);
    __syncthreads();                   // all waves done reading `cur` before DMA reuses it
    cur ^= 1;
  }

  // Epilogue: C layout -> VGPR v: lanes 0-15 hold (M=v, N=lane), lanes 16-31 hold (M=v+8).
#pragma unroll
  for (int mi = 0; mi < 2; ++mi) {
#pragma unroll
    for (int v = 0; v < 8; ++v) {
      int gr = rowbase + mi * 16 + v + hi * 8;
      float s = 0.0f;
#pragma unroll
      for (int ni = 0; ni < 4; ++ni) {
        int gc = colbase + ni * 16 + lp;
        float val = acc[mi][ni][v];
        if (gr != gc) s += __expf(val * INV_T);   // mask the j==i diagonal
      }
      for (int off = 1; off < 16; off <<= 1) s += __shfl_xor(s, off, 32);
      if (lp == 0) atomicAdd(&rowsum[h * NROWS + gr], s);
    }
  }
}

// ---------------- 4) finalize ----------------
__global__ void finalize_kernel(const float* __restrict__ rowsum, const float* __restrict__ pos,
                                float* __restrict__ out) {
  int idx = blockIdx.x * blockDim.x + threadIdx.x;
  float v = 0.0f;
  for (int i = idx; i < 2 * NROWS; i += gridDim.x * blockDim.x)
    v += logf(rowsum[i]) - pos[i] * INV_T;
  __shared__ float red[256];
  int t = threadIdx.x;
  red[t] = v;
  __syncthreads();
  for (int k = 128; k > 0; k >>= 1) { if (t < k) red[t] += red[t + k]; __syncthreads(); }
  if (t == 0) atomicAdd(out, red[0] * (1.0f / NROWS));   // divide by 2B = 8192
}

extern "C" void kernel_launch(void* const* d_in, const int* in_sizes, int n_in,
                              void* d_out, int out_size, void* d_ws, size_t ws_size,
                              hipStream_t stream) {
  const float* x1 = (const float*)d_in[0];
  const float* x2 = (const float*)d_in[1];
  const float* z1 = (const float*)d_in[2];
  const float* z2 = (const float*)d_in[3];
  float* out = (float*)d_out;

  // workspace: Nh (2*8192*512 f16 = 16 MB) | rowsum (16384 f32) | pos (16384 f32)
  _Float16* Nh  = (_Float16*)d_ws;
  float* rowsum = (float*)((char*)d_ws + (size_t)2 * NROWS * DDIM * sizeof(_Float16));
  float* pos    = rowsum + 2 * NROWS;

  (void)hipMemsetAsync(rowsum, 0, (size_t)2 * NROWS * sizeof(float), stream);
  (void)hipMemsetAsync(out, 0, sizeof(float), stream);

  normalize_kernel<<<2 * NROWS, 256, 0, stream>>>(x1, x2, z1, z2, Nh);
  pos_kernel<<<2 * NROWS, 256, 0, stream>>>(Nh, pos);

  dim3 grid(NROWS / BN, NROWS / BM, 2);   // 64 x 64 x 2 blocks, 256 threads each
  gemm_exp_kernel<<<grid, 256, 0, stream>>>(Nh, rowsum);

  finalize_kernel<<<16, 256, 0, stream>>>(rowsum, pos, out);
}
